// MultiheadDiffAttn_14809047236974
// MI455X (gfx1250) — matrix operand order repaired and checked by reading, not compiled, not run on hardware
//
#include <hip/hip_runtime.h>
#include <hip/hip_bf16.h>
#include <math.h>

// ---------------------------------------------------------------------------
// MI455X (gfx1250) differential attention, full pipeline in f16 WMMA with
// fp32 accumulation.  wave32, 16x16x32 f16 WMMA throughout.
// ---------------------------------------------------------------------------

typedef _Float16 half8 __attribute__((ext_vector_type(8), aligned(16)));
typedef _Float16 v16h  __attribute__((ext_vector_type(16)));
typedef float    v8f   __attribute__((ext_vector_type(8)));

// LAMBDA_INIT = 0.8 - 0.6*exp(-0.3*12)
static constexpr float LAMBDA_INIT = 0.783605766531624f;

__device__ inline v16h make_v16(half8 lo, half8 hi) {
    v16h r;
#pragma unroll
    for (int i = 0; i < 8; ++i) { r[i] = lo[i]; r[i + 8] = hi[i]; }
    return r;
}

// ---------------------------------------------------------------------------
// fp32 -> fp16 conversion
// ---------------------------------------------------------------------------
__global__ void cvt_f32_f16(const float* __restrict__ in,
                            _Float16* __restrict__ out, int n) {
    int i = blockIdx.x * blockDim.x + threadIdx.x;
    if (i < n) out[i] = (_Float16)in[i];
}

// ---------------------------------------------------------------------------
// C[M,N] = A[M,K] * B[N,K]^T   (torch Linear layout: both K-contiguous)
// 128x128 block tile, 8 waves, each wave: 2(M) x 4(N) tiles of 16x16.
// ---------------------------------------------------------------------------
template <bool OUT_F32>
__global__ void __launch_bounds__(256)
gemm_f16_nt(const _Float16* __restrict__ A, const _Float16* __restrict__ B,
            void* __restrict__ C, int M, int N, int K) {
    const int lane    = threadIdx.x & 31;
    const int wave    = threadIdx.x >> 5;     // 0..7
    const int lrow    = lane & 15;
    const int halfSel = (lane >> 4) & 1;
    const int wm = blockIdx.x * 128 + (wave & 3) * 32;  // 2 m-tiles
    const int wn = blockIdx.y * 128 + (wave >> 2) * 64; // 4 n-tiles

    v8f acc[2][4];
#pragma unroll
    for (int i = 0; i < 2; ++i)
#pragma unroll
        for (int j = 0; j < 4; ++j)
#pragma unroll
            for (int e = 0; e < 8; ++e) acc[i][j][e] = 0.0f;

    for (int k0 = 0; k0 < K; k0 += 32) {
        // A fragments: lane = row; K chunks {0..7,16..23} / {8..15,24..31}
        v16h af[2];
#pragma unroll
        for (int mt = 0; mt < 2; ++mt) {
            const _Float16* p =
                A + (size_t)(wm + mt * 16 + lrow) * K + k0 + halfSel * 8;
            af[mt] = make_v16(*(const half8*)p, *(const half8*)(p + 16));
        }
        // B fragments: lane = column (row of B-operand-source); K chunks {0..7,16..23} / {8..15,24..31}
        v16h bf[4];
#pragma unroll
        for (int nt = 0; nt < 4; ++nt) {
            const _Float16* p =
                B + (size_t)(wn + nt * 16 + lrow) * K + k0 + halfSel * 8;
            bf[nt] = make_v16(*(const half8*)p, *(const half8*)(p + 16));
        }
#pragma unroll
        for (int mt = 0; mt < 2; ++mt)
#pragma unroll
            for (int nt = 0; nt < 4; ++nt)
                acc[mt][nt] = __builtin_amdgcn_wmma_f32_16x16x32_f16(
                    false, af[mt], false, bf[nt], (short)0, acc[mt][nt],
                    false, false);
    }

    // C layout: VGPR e <-> row e + halfSel*8, column = lrow
#pragma unroll
    for (int mt = 0; mt < 2; ++mt)
#pragma unroll
        for (int nt = 0; nt < 4; ++nt)
#pragma unroll
            for (int e = 0; e < 8; ++e) {
                int m = wm + mt * 16 + e + halfSel * 8;
                int n = wn + nt * 16 + lrow;
                float v = acc[mt][nt][e];
                if (OUT_F32)
                    ((float*)C)[(size_t)m * N + n] = v;
                else
                    ((_Float16*)C)[(size_t)m * N + n] = (_Float16)v;
            }
}

// ---------------------------------------------------------------------------
// vh [4096][1024] (f16)  ->  vt [b][h][dim=128][token=2048] (f16)
// so the P.V B-operand (column = value dim, contiguous tokens) is contiguous.
// ---------------------------------------------------------------------------
__global__ void transpose_v(const _Float16* __restrict__ vh,
                            _Float16* __restrict__ vt) {
    int idx = blockIdx.x * blockDim.x + threadIdx.x;  // 2*8*128*2048 = 4M
    int t = idx & 2047;
    int n = (idx >> 11) & 127;
    int h = (idx >> 18) & 7;
    int b = idx >> 21;
    vt[idx] = vh[((size_t)(b * 2048 + t)) * 1024 + h * 128 + n];
}

// ---------------------------------------------------------------------------
// lambda = exp(dot(lq1,lk1)) - exp(dot(lq2,lk2)) + LAMBDA_INIT
// ---------------------------------------------------------------------------
__global__ void lambda_kernel(const float* __restrict__ lq1,
                              const float* __restrict__ lk1,
                              const float* __restrict__ lq2,
                              const float* __restrict__ lk2,
                              float* __restrict__ lam) {
    if (threadIdx.x == 0) {
        float s1 = 0.f, s2 = 0.f;
        for (int i = 0; i < 64; ++i) {
            s1 += lq1[i] * lk1[i];
            s2 += lq2[i] * lk2[i];
        }
        *lam = expf(s1) - expf(s2) + LAMBDA_INIT;
    }
}

// ---------------------------------------------------------------------------
// Fused differential flash attention + RMSNorm.
// Grid: (B*H=16, S/128=16), 256 threads (8 waves); each wave owns a 16-row
// query strip.  For g in {0,1}: online softmax over 16 key blocks, P staged
// through padded per-wave LDS to re-shape C-layout -> A-fragment layout.
// Output: O0 - lam*O1, RMSNorm over 2d=128, *(1-LAMBDA_INIT), stored f16.
// ---------------------------------------------------------------------------
__global__ void __launch_bounds__(256)
diff_attn(const _Float16* __restrict__ qh, const _Float16* __restrict__ kh,
          const _Float16* __restrict__ vt, _Float16* __restrict__ on,
          const float* __restrict__ lamPtr) {
    constexpr int S = 2048, E = 1024, D = 64, D2 = 128;
    const float scaling = 0.125f;  // 64^-0.5
    const int lane    = threadIdx.x & 31;
    const int wave    = threadIdx.x >> 5;
    const int lrow    = lane & 15;
    const int halfSel = (lane >> 4) & 1;
    const int bh = blockIdx.x;  // b*8+h
    const int h  = bh & 7;
    const int b  = bh >> 3;
    const int qrow0 = blockIdx.y * 128 + wave * 16;

    const float lam = *lamPtr;

    __shared__ __align__(16) _Float16 pbuf[8][16][136];  // padded: 272B row stride
    _Float16(*pw)[136] = pbuf[wave];

    v8f oF[8];
#pragma unroll
    for (int t = 0; t < 8; ++t)
#pragma unroll
        for (int e = 0; e < 8; ++e) oF[t][e] = 0.f;

    for (int g = 0; g < 2; ++g) {
        // Preload Q A-fragments (K = head dim, 2 steps of 32)
        const _Float16* qbase =
            qh + ((size_t)(b * S) + qrow0 + lrow) * E + h * D2 + g * D;
        v16h qf[2];
#pragma unroll
        for (int ks = 0; ks < 2; ++ks) {
            const _Float16* p = qbase + ks * 32 + halfSel * 8;
            qf[ks] = make_v16(*(const half8*)p, *(const half8*)(p + 16));
        }

        float mrun[8], lrun[8];
        v8f oc[8];
#pragma unroll
        for (int e = 0; e < 8; ++e) { mrun[e] = -3.0e38f; lrun[e] = 0.f; }
#pragma unroll
        for (int t = 0; t < 8; ++t)
#pragma unroll
            for (int e = 0; e < 8; ++e) oc[t][e] = 0.f;

        for (int kb = 0; kb < 16; ++kb) {
            // ---- scores strip S = Q K^T (16 x 128) ----
            v8f s[8];
#pragma unroll
            for (int t = 0; t < 8; ++t)
#pragma unroll
                for (int e = 0; e < 8; ++e) s[t][e] = 0.f;
#pragma unroll
            for (int ks = 0; ks < 2; ++ks) {
#pragma unroll
                for (int nt = 0; nt < 8; ++nt) {
                    const _Float16* p =
                        kh + ((size_t)(b * S) + kb * 128 + nt * 16 + lrow) * E +
                        h * D2 + g * D + ks * 32 + halfSel * 8;
                    v16h bf_ = make_v16(*(const half8*)p, *(const half8*)(p + 16));
                    s[nt] = __builtin_amdgcn_wmma_f32_16x16x32_f16(
                        false, qf[ks], false, bf_, (short)0, s[nt], false, false);
                }
            }
            // ---- online softmax over this key block ----
#pragma unroll
            for (int e = 0; e < 8; ++e) {
                float mx = -3.0e38f;
#pragma unroll
                for (int t = 0; t < 8; ++t) {
                    float v = s[t][e] * scaling;
                    s[t][e] = v;
                    mx = fmaxf(mx, v);
                }
#pragma unroll
                for (int m_ = 1; m_ < 16; m_ <<= 1)
                    mx = fmaxf(mx, __shfl_xor(mx, m_, 32));
                mx = fmaxf(mx, mrun[e]);
                float ps = 0.f;
#pragma unroll
                for (int t = 0; t < 8; ++t) {
                    float pv = __expf(s[t][e] - mx);
                    s[t][e] = pv;
                    ps += pv;
                }
#pragma unroll
                for (int m_ = 1; m_ < 16; m_ <<= 1)
                    ps += __shfl_xor(ps, m_, 32);
                float sc = __expf(mrun[e] - mx);
                lrun[e] = lrun[e] * sc + ps;
                mrun[e] = mx;
#pragma unroll
                for (int t = 0; t < 8; ++t) oc[t][e] *= sc;
            }
            // ---- stage P strip to LDS (C layout -> row major f16) ----
#pragma unroll
            for (int t = 0; t < 8; ++t)
#pragma unroll
                for (int e = 0; e < 8; ++e)
                    pw[e + halfSel * 8][t * 16 + lrow] = (_Float16)s[t][e];
            // ---- O += P V  (K = 128 keys, 4 steps of 32) ----
#pragma unroll
            for (int kv = 0; kv < 4; ++kv) {
                const _Float16* pr = &pw[lrow][kv * 32 + halfSel * 8];
                v16h pf = make_v16(*(const half8*)pr, *(const half8*)(pr + 16));
#pragma unroll
                for (int t = 0; t < 8; ++t) {
                    const _Float16* vp =
                        vt + ((size_t)bh * D2 + t * 16 + lrow) * S + kb * 128 +
                        kv * 32 + halfSel * 8;
                    v16h vf = make_v16(*(const half8*)vp, *(const half8*)(vp + 16));
                    oc[t] = __builtin_amdgcn_wmma_f32_16x16x32_f16(
                        false, pf, false, vf, (short)0, oc[t], false, false);
                }
            }
        }
        // ---- normalize & differential combine ----
        float coeff = (g == 0) ? 1.0f : -lam;
#pragma unroll
        for (int e = 0; e < 8; ++e) {
            float inv = coeff / lrun[e];
#pragma unroll
            for (int t = 0; t < 8; ++t) oF[t][e] += oc[t][e] * inv;
        }
    }

    // ---- RMSNorm over 2d=128 per row, * (1 - LAMBDA_INIT), store f16 ----
#pragma unroll
    for (int e = 0; e < 8; ++e) {
        float ss = 0.f;
#pragma unroll
        for (int t = 0; t < 8; ++t) ss += oF[t][e] * oF[t][e];
#pragma unroll
        for (int m_ = 1; m_ < 16; m_ <<= 1) ss += __shfl_xor(ss, m_, 32);
        float r = rsqrtf(ss * (1.0f / 128.0f) + 1e-5f) * (1.0f - LAMBDA_INIT);
        int tok = qrow0 + e + halfSel * 8;
#pragma unroll
        for (int t = 0; t < 8; ++t)
            on[((size_t)(b * S) + tok) * E + h * D2 + t * 16 + lrow] =
                (_Float16)(oF[t][e] * r);
    }
}

// ---------------------------------------------------------------------------
extern "C" void kernel_launch(void* const* d_in, const int* in_sizes, int n_in,
                              void* d_out, int out_size, void* d_ws,
                              size_t ws_size, hipStream_t stream) {
    (void)in_sizes; (void)n_in; (void)out_size; (void)ws_size;
    const float* x   = (const float*)d_in[0];
    const float* Wq  = (const float*)d_in[1];
    const float* Wk  = (const float*)d_in[2];
    const float* Wv  = (const float*)d_in[3];
    const float* Wo  = (const float*)d_in[4];
    const float* lq1 = (const float*)d_in[5];
    const float* lk1 = (const float*)d_in[6];
    const float* lq2 = (const float*)d_in[7];
    const float* lk2 = (const float*)d_in[8];

    char* ws = (char*)d_ws;
    const size_t MB = 1ull << 20;
    _Float16* xh  = (_Float16*)(ws);             // 8 MB  : x f16
    _Float16* Wqh = (_Float16*)(ws + 8  * MB);   // 2 MB
    _Float16* Wkh = (_Float16*)(ws + 10 * MB);   // 2 MB
    _Float16* Wvh = (_Float16*)(ws + 12 * MB);   // 2 MB
    _Float16* Woh = (_Float16*)(ws + 14 * MB);   // 2 MB
    _Float16* qh  = (_Float16*)(ws + 16 * MB);   // 8 MB
    _Float16* kh  = (_Float16*)(ws + 24 * MB);   // 8 MB
    _Float16* vh  = (_Float16*)(ws + 32 * MB);   // 8 MB
    _Float16* vt  = (_Float16*)(ws + 40 * MB);   // 8 MB  : V transposed
    _Float16* onb = (_Float16*)(ws + 48 * MB);   // 8 MB  : normed attn out
    float*    lam = (float*)(ws + 56 * MB);

    const int NX = 4096 * 1024, NW = 1024 * 1024;
    cvt_f32_f16<<<(NX + 255) / 256, 256, 0, stream>>>(x, xh, NX);
    cvt_f32_f16<<<(NW + 255) / 256, 256, 0, stream>>>(Wq, Wqh, NW);
    cvt_f32_f16<<<(NW + 255) / 256, 256, 0, stream>>>(Wk, Wkh, NW);
    cvt_f32_f16<<<(NW + 255) / 256, 256, 0, stream>>>(Wv, Wvh, NW);
    cvt_f32_f16<<<(NW + 255) / 256, 256, 0, stream>>>(Wo, Woh, NW);

    dim3 g128(4096 / 128, 1024 / 128);  // (32, 8)
    gemm_f16_nt<false><<<g128, 256, 0, stream>>>(xh, Wqh, qh, 4096, 1024, 1024);
    gemm_f16_nt<false><<<g128, 256, 0, stream>>>(xh, Wkh, kh, 4096, 1024, 1024);
    gemm_f16_nt<false><<<g128, 256, 0, stream>>>(xh, Wvh, vh, 4096, 1024, 1024);

    transpose_v<<<(2 * 8 * 128 * 2048) / 256, 256, 0, stream>>>(vh, vt);
    lambda_kernel<<<1, 32, 0, stream>>>(lq1, lk1, lq2, lk2, lam);

    dim3 ga(16, 16);  // (B*H, S/128)
    diff_attn<<<ga, 256, 0, stream>>>(qh, kh, vt, onb, lam);

    gemm_f16_nt<true><<<g128, 256, 0, stream>>>(onb, Woh, d_out, 4096, 1024, 1024);
}
